// VectorizedTopKMoE_37177236914640
// MI455X (gfx1250) — matrix operand order repaired and checked by reading, not compiled, not run on hardware
//
#include <hip/hip_runtime.h>

#define N_TOKENS  4096
#define HIDDEN    1024
#define N_EXPERTS 8
#define EXPANDED  2048

typedef __attribute__((ext_vector_type(16))) __bf16 v16bf;
typedef __attribute__((ext_vector_type(8)))  float  v8f;
typedef unsigned int u32x4 __attribute__((ext_vector_type(4)));

union FragBF { v16bf v; u32x4 q[2]; };

__device__ __forceinline__ unsigned short f2bf(float f) {
  unsigned int u = __builtin_bit_cast(unsigned int, f);
  unsigned int r = u + 0x7FFFu + ((u >> 16) & 1u);  // round-to-nearest-even
  return (unsigned short)(r >> 16);
}

__device__ __forceinline__ float gelu_exact(float x) {
  return 0.5f * x * (1.0f + erff(x * 0.70710678118654752f));
}

__device__ __forceinline__ v8f v8f_zero() {
  v8f z;
#pragma unroll
  for (int i = 0; i < 8; ++i) z[i] = 0.0f;
  return z;
}

// Async global->LDS copy of 16B per lane, tracked with ASYNCcnt (bypasses VGPRs).
__device__ __forceinline__ void async_copy_b128(unsigned lds_off, const void* gaddr) {
  asm volatile("global_load_async_to_lds_b128 %0, %1, off"
               :: "v"(lds_off), "v"(gaddr)
               : "memory");
}

__device__ __forceinline__ void wait_asynccnt0() {
  asm volatile("s_wait_asynccnt 0x0" ::: "memory");
}

// ---------------------------------------------------------------- utilities
__global__ void zero_f32_kernel(float* __restrict__ p, int n) {
  for (int i = blockIdx.x * blockDim.x + threadIdx.x; i < n; i += gridDim.x * blockDim.x)
    p[i] = 0.0f;
}

__global__ void zero_u32_kernel(unsigned int* __restrict__ p, int n) {
  for (int i = blockIdx.x * blockDim.x + threadIdx.x; i < n; i += gridDim.x * blockDim.x)
    p[i] = 0u;
}

__global__ void cvt_bf16_kernel(const float* __restrict__ in, unsigned short* __restrict__ out, int n) {
  for (int i = blockIdx.x * blockDim.x + threadIdx.x; i < n; i += gridDim.x * blockDim.x)
    out[i] = f2bf(in[i]);
}

// Transpose [K][N] fp32 -> [N][K] bf16 (per expert via blockIdx.z), 32x32 tiles.
__global__ void transpose_cvt_kernel(const float* __restrict__ in, unsigned short* __restrict__ out,
                                     int K, int N) {
  __shared__ float tile[32][33];
  const size_t base = (size_t)blockIdx.z * K * N;
  const int n = blockIdx.x * 32 + threadIdx.x;
  const int k0 = blockIdx.y * 32;
#pragma unroll
  for (int j = 0; j < 4; ++j)
    tile[threadIdx.y + 8 * j][threadIdx.x] = in[base + (size_t)(k0 + threadIdx.y + 8 * j) * N + n];
  __syncthreads();
  const int kOut = k0 + threadIdx.x;
#pragma unroll
  for (int j = 0; j < 4; ++j) {
    const int nOut = blockIdx.x * 32 + threadIdx.y + 8 * j;
    out[base + (size_t)nOut * K + kOut] = f2bf(tile[threadIdx.x][threadIdx.y + 8 * j]);
  }
}

// ------------------------------------------------------------------- router
// One wave32 per token: 8 logits, top-2 softmax, build compacted expert lists.
__global__ void router_kernel(const float* __restrict__ tokens,
                              const float* __restrict__ rw,
                              const float* __restrict__ rb,
                              int* __restrict__ counts,
                              int* __restrict__ idxList,
                              float* __restrict__ gateList) {
  const int wave = threadIdx.x >> 5;
  const int lane = threadIdx.x & 31;
  const int t = blockIdx.x * 8 + wave;
  const float* tok = tokens + (size_t)t * HIDDEN;
  float acc[N_EXPERTS];
#pragma unroll
  for (int e = 0; e < N_EXPERTS; ++e) acc[e] = 0.0f;
  for (int k = lane; k < HIDDEN; k += 32) {
    const float x = tok[k];
#pragma unroll
    for (int e = 0; e < N_EXPERTS; ++e) acc[e] += x * rw[e * HIDDEN + k];
  }
#pragma unroll
  for (int e = 0; e < N_EXPERTS; ++e)
    for (int off = 16; off; off >>= 1) acc[e] += __shfl_xor(acc[e], off, 32);
  if (lane == 0) {
#pragma unroll
    for (int e = 0; e < N_EXPERTS; ++e) acc[e] += rb[e];
    int i1 = 0; float s1 = acc[0];
#pragma unroll
    for (int e = 1; e < N_EXPERTS; ++e) if (acc[e] > s1) { s1 = acc[e]; i1 = e; }
    int i2 = -1; float s2 = -3.4e38f;
#pragma unroll
    for (int e = 0; e < N_EXPERTS; ++e) if (e != i1 && acc[e] > s2) { s2 = acc[e]; i2 = e; }
    const float e2 = expf(s2 - s1);
    const float inv = 1.0f / (1.0f + e2);
    const float p1 = inv, p2 = e2 * inv;
    int pos = atomicAdd(&counts[i1], 1);
    idxList[i1 * N_TOKENS + pos] = t; gateList[i1 * N_TOKENS + pos] = p1;
    pos = atomicAdd(&counts[i2], 1);
    idxList[i2 * N_TOKENS + pos] = t; gateList[i2 * N_TOKENS + pos] = p2;
  }
}

// --------------------------------------------------------------------- GEMM
// 128x128 block tile, BK=32, 8 waves (4x2), wave tile 32x64 (2x4 WMMA frags).
// Staging: async global->LDS copies (ASYNCcnt) double-buffered against WMMAs.
// GEMM1: H = gelu(gather(X) @ W1t^T + b1) -> bf16 (compacted rows)
// GEMM2: out[tok] += gate * (H @ W2t^T + b2)  (scatter)
template <bool GEMM1>
__global__ __launch_bounds__(256) void moe_gemm_kernel(
    const unsigned short* __restrict__ A,     // bf16: X [4096][K] or H [*][K]
    const unsigned short* __restrict__ Bt,    // bf16 transposed weights [N][K]
    const float* __restrict__ bias,           // [N]
    const int* __restrict__ counts,
    const int* __restrict__ idxList,          // this expert's slice [4096]
    const float* __restrict__ gateList,       // this expert's slice [4096]
    unsigned short* __restrict__ Hout,        // GEMM1 output
    float* __restrict__ out,                  // GEMM2 output
    int N, int K, int expert) {
  const int count = counts[expert];
  const int rowA0 = blockIdx.y * 128;
  if (rowA0 >= count) return;  // routed-token sparsity: skip empty blocks

  __shared__ __align__(16) unsigned short As[2][128][40];  // +8 bf16 pad per row
  __shared__ __align__(16) unsigned short Bs[2][128][40];

  const int t = threadIdx.x;
  const int lane = t & 31, wave = t >> 5;
  const int waveM = wave >> 1, waveN = wave & 1;
  const int half = lane >> 4, r16 = lane & 15;
  const int rowB0 = blockIdx.x * 128;

  // Per-thread source row offsets for the two 16B copy slots (gather for GEMM1).
  size_t aRowOff[2];
#pragma unroll
  for (int i = 0; i < 2; ++i) {
    const int idx = i * 256 + t;
    const int r = idx >> 2;
    const int srcRow = GEMM1 ? idxList[rowA0 + r] : (rowA0 + r);
    aRowOff[i] = (size_t)srcRow * K;
  }

  auto load_tiles = [&](int kt, int buf) {
    const int k0 = kt * 32;
#pragma unroll
    for (int i = 0; i < 2; ++i) {
      const int idx = i * 256 + t;
      const int r = idx >> 2;
      const int c = (idx & 3) * 8;
      // LDS-aperture flat address truncates to the LDS byte offset (ISA 10.2).
      const unsigned ldsA = (unsigned)(uintptr_t)&As[buf][r][c];
      const unsigned ldsB = (unsigned)(uintptr_t)&Bs[buf][r][c];
      async_copy_b128(ldsA, A + aRowOff[i] + k0 + c);
      async_copy_b128(ldsB, Bt + (size_t)(rowB0 + r) * K + k0 + c);
    }
  };

  v8f acc[2][4];
#pragma unroll
  for (int mr = 0; mr < 2; ++mr)
#pragma unroll
    for (int nr = 0; nr < 4; ++nr) acc[mr][nr] = v8f_zero();

  const int KT = K >> 5;
  load_tiles(0, 0);
  wait_asynccnt0();
  __syncthreads();
  for (int kt = 0; kt < KT; ++kt) {
    const int buf = kt & 1;
    if (kt + 1 < KT) load_tiles(kt + 1, buf ^ 1);  // async fill of other buffer

    FragBF fa[2], fb[4];
    // A frag (16x32 bf16): lanes<16 hold K{0..7,16..23}, lanes>=16 hold K{8..15,24..31}
#pragma unroll
    for (int mr = 0; mr < 2; ++mr) {
      const int r = waveM * 32 + mr * 16 + r16;
      fa[mr].q[0] = *reinterpret_cast<const u32x4*>(&As[buf][r][half * 8]);
      fa[mr].q[1] = *reinterpret_cast<const u32x4*>(&As[buf][r][16 + half * 8]);
    }
    // B frag (32x16 bf16, stored N-major): lanes<16 hold K{0..7,16..23}, lanes>=16 hold K{8..15,24..31}
#pragma unroll
    for (int nr = 0; nr < 4; ++nr) {
      const int r = waveN * 64 + nr * 16 + r16;
      fb[nr].q[0] = *reinterpret_cast<const u32x4*>(&Bs[buf][r][half * 8]);
      fb[nr].q[1] = *reinterpret_cast<const u32x4*>(&Bs[buf][r][16 + half * 8]);
    }
#pragma unroll
    for (int mr = 0; mr < 2; ++mr)
#pragma unroll
      for (int nr = 0; nr < 4; ++nr)
        acc[mr][nr] = __builtin_amdgcn_wmma_f32_16x16x32_bf16(
            false, fa[mr].v, false, fb[nr].v, (short)0, acc[mr][nr], false, false);

    wait_asynccnt0();   // next tile landed in LDS
    __syncthreads();    // all waves' copies visible before buffer swap
  }

  // Epilogue. C layout: VGPR i holds M = i (+8 for lanes>=16), N = lane&15.
  float bcol[4];
#pragma unroll
  for (int nr = 0; nr < 4; ++nr) bcol[nr] = bias[rowB0 + waveN * 64 + nr * 16 + r16];

#pragma unroll
  for (int mr = 0; mr < 2; ++mr) {
#pragma unroll
    for (int i = 0; i < 8; ++i) {
      const int row = rowA0 + waveM * 32 + mr * 16 + half * 8 + i;
      if (GEMM1) {
#pragma unroll
        for (int nr = 0; nr < 4; ++nr) {
          const int col = rowB0 + waveN * 64 + nr * 16 + r16;
          float v = acc[mr][nr][i] + bcol[nr];
          v = gelu_exact(v);
          Hout[(size_t)row * N + col] = f2bf(v);
        }
      } else {
        if (row < count) {
          const int tok = idxList[row];
          const float g = gateList[row];
#pragma unroll
          for (int nr = 0; nr < 4; ++nr) {
            const int col = rowB0 + waveN * 64 + nr * 16 + r16;
            out[(size_t)tok * N + col] += g * (acc[mr][nr][i] + bcol[nr]);
          }
        }
      }
    }
  }
}

// --------------------------------------------------------------------- host
extern "C" void kernel_launch(void* const* d_in, const int* in_sizes, int n_in,
                              void* d_out, int out_size, void* d_ws, size_t ws_size,
                              hipStream_t stream) {
  const float* tokens   = (const float*)d_in[0];
  const float* router_w = (const float*)d_in[1];
  const float* router_b = (const float*)d_in[2];
  const float* w1       = (const float*)d_in[3];
  const float* b1       = (const float*)d_in[4];
  const float* w2       = (const float*)d_in[5];
  const float* b2       = (const float*)d_in[6];
  float* out = (float*)d_out;

  char* ws = (char*)d_ws;
  size_t off = 0;
  auto alloc = [&](size_t bytes) -> void* {
    void* p = ws + off;
    off = (off + bytes + 255) & ~(size_t)255;
    return p;
  };
  unsigned short* Xbf = (unsigned short*)alloc((size_t)N_TOKENS * HIDDEN * 2);
  unsigned short* W1t = (unsigned short*)alloc((size_t)N_EXPERTS * HIDDEN * EXPANDED * 2);
  unsigned short* W2t = (unsigned short*)alloc((size_t)N_EXPERTS * EXPANDED * HIDDEN * 2);
  unsigned short* Hbf = (unsigned short*)alloc((size_t)N_TOKENS * EXPANDED * 2);
  int*   idxList  = (int*)alloc((size_t)N_EXPERTS * N_TOKENS * 4);
  float* gateList = (float*)alloc((size_t)N_EXPERTS * N_TOKENS * 4);
  int*   counts   = (int*)alloc((size_t)N_EXPERTS * 4);

  // Zero routing state (idxList, gateList, counts are contiguous 256B-aligned
  // regions in that order) and the output accumulator.
  zero_u32_kernel<<<64, 256, 0, stream>>>((unsigned int*)idxList,
                                          2 * N_EXPERTS * N_TOKENS + N_EXPERTS);
  zero_f32_kernel<<<1024, 256, 0, stream>>>(out, N_TOKENS * HIDDEN);

  // Precision conversion + weight transposition (one-time, bandwidth-bound).
  cvt_bf16_kernel<<<2048, 256, 0, stream>>>(tokens, Xbf, N_TOKENS * HIDDEN);
  dim3 tb(32, 8);
  transpose_cvt_kernel<<<dim3(EXPANDED / 32, HIDDEN / 32, N_EXPERTS), tb, 0, stream>>>(
      w1, W1t, HIDDEN, EXPANDED);
  transpose_cvt_kernel<<<dim3(HIDDEN / 32, EXPANDED / 32, N_EXPERTS), tb, 0, stream>>>(
      w2, W2t, EXPANDED, HIDDEN);

  // Router + expert list construction.
  router_kernel<<<N_TOKENS / 8, 256, 0, stream>>>(tokens, router_w, router_b,
                                                  counts, idxList, gateList);

  // Per-expert fused MLP over routed tokens only (blocks past count exit early).
  for (int e = 0; e < N_EXPERTS; ++e) {
    moe_gemm_kernel<true><<<dim3(EXPANDED / 128, N_TOKENS / 128), 256, 0, stream>>>(
        Xbf, W1t + (size_t)e * HIDDEN * EXPANDED, b1 + (size_t)e * EXPANDED,
        counts, idxList + e * N_TOKENS, gateList + e * N_TOKENS,
        Hbf, nullptr, EXPANDED, HIDDEN, e);
    moe_gemm_kernel<false><<<dim3(HIDDEN / 128, N_TOKENS / 128), 256, 0, stream>>>(
        Hbf, W2t + (size_t)e * EXPANDED * HIDDEN, b2 + (size_t)e * HIDDEN,
        counts, idxList + e * N_TOKENS, gateList + e * N_TOKENS,
        nullptr, out, HIDDEN, EXPANDED, e);
  }
}
